// CosGaussianKernelDiagonal_85959475462553
// MI455X (gfx1250) — compile-verified, matrix operand order checked by reading, not run on hardware
//
#include <hip/hip_runtime.h>
#include <math.h>

// CDNA5 / gfx1250 wave32 WMMA kernel for:
//   out[b,n,m] = cos(x@mu[n] - y@mu[m]) * exp(-0.5*max(|x_n|^2 + |y_m|^2 - 2 x_n.y_m, 0))
// with x_,y_ diagonal-scaled by exp(0.5*logs2).
//
// The 2*xy GEMM (K=64) is done on the matrix cores via bf16 hi/lo split
// (hi*hi + hi*lo + lo*hi ~ fp32 accuracy), v_wmma_f32_16x16x32_bf16.
// Output-store-bound floor ~ 256MB / 23.3TB/s ~= 11us; epilogue fused.

typedef __attribute__((ext_vector_type(16))) __bf16 v16bf;
typedef __attribute__((ext_vector_type(8)))  __bf16 bf16x8;
typedef __attribute__((ext_vector_type(8)))  float  v8f;

#define TILE  128
#define D_DIM 64
#define KP    72   // padded LDS K-stride (bf16 units): 144B rows, 16B aligned

struct SMem {
  __bf16 xhi[TILE * KP];
  __bf16 xlo[TILE * KP];
  __bf16 yhi[TILE * KP];
  __bf16 ylo[TILE * KP];
  float  mx[TILE], x2[TILE];   // per x-row stats (n)
  float  my[TILE], y2[TILE];   // per y-row stats (m)
  float  mu[D_DIM], ssq[D_DIM];
};

// Build a 16-element bf16 WMMA fragment from two aligned 16B LDS chunks.
// Per ISA 16-bit A/B layout: elements 0..7 = K[kb..kb+7], 8..15 = K[kb+16..kb+23].
__device__ __forceinline__ v16bf load_frag(const __bf16* base) {
  union { v16bf v; bf16x8 h[2]; } u;
  u.h[0] = *reinterpret_cast<const bf16x8*>(base);
  u.h[1] = *reinterpret_cast<const bf16x8*>(base + 16);
  return u.v;
}

__global__ __launch_bounds__(256)
void cosgauss_wmma_kernel(const float* __restrict__ x,
                          const float* __restrict__ y,
                          const float* __restrict__ mu,
                          const float* __restrict__ logs2,
                          float* __restrict__ out,
                          int N, int M) {
  __shared__ SMem sm;
  const int tid   = threadIdx.x;
  const int b     = blockIdx.z;
  const int ntile = blockIdx.y;           // rows  (n, from x)
  const int mtile = blockIdx.x;           // cols  (m, from y) — contiguous in out

  // ---- Preload mu and diagonal scale s = exp(0.5*logs2) ----
  if (tid < D_DIM) {
    sm.mu[tid]  = mu[tid];
    sm.ssq[tid] = __expf(0.5f * logs2[tid]);
  }
  __syncthreads();

  // ---- Phase 1: load tile rows, scale, bf16 hi/lo split, row stats ----
  {
    const bool isX = (tid < TILE);
    const int  r   = isX ? tid : (tid - TILE);
    const float* src = isX
        ? (x + ((size_t)b * N + (size_t)ntile * TILE + r) * D_DIM)
        : (y + ((size_t)b * M + (size_t)mtile * TILE + r) * D_DIM);
    __bf16* hi = isX ? &sm.xhi[r * KP] : &sm.yhi[r * KP];
    __bf16* lo = isX ? &sm.xlo[r * KP] : &sm.ylo[r * KP];

    const float4* s4  = reinterpret_cast<const float4*>(src);
    const float4* mu4 = reinterpret_cast<const float4*>(sm.mu);
    const float4* sc4 = reinterpret_cast<const float4*>(sm.ssq);

    float mdot = 0.f, sq = 0.f;
    #pragma unroll
    for (int i = 0; i < D_DIM / 4; ++i) {
      float4 v = s4[i];
      float4 m = mu4[i];
      float4 s = sc4[i];
      float raw[4] = {v.x, v.y, v.z, v.w};
      float mm[4]  = {m.x, m.y, m.z, m.w};
      float ss[4]  = {s.x, s.y, s.z, s.w};
      #pragma unroll
      for (int c = 0; c < 4; ++c) {
        const int d  = i * 4 + c;
        const float xs = raw[c] * ss[c];
        mdot = fmaf(raw[c], mm[c], mdot);   // raw x @ mu   (cos term)
        sq   = fmaf(xs, xs, sq);            // |scaled|^2
        const __bf16 h = (__bf16)xs;
        hi[d] = h;
        lo[d] = (__bf16)(xs - (float)h);
      }
    }
    if (isX) { sm.mx[r] = mdot; sm.x2[r] = sq; }
    else     { sm.my[r] = mdot; sm.y2[r] = sq; }
  }
  __syncthreads();

  // ---- Phase 2: WMMA GEMM. 8 waves in 2x4 grid; each wave: 64x32 = 4x2 tiles ----
  const int lane = tid & 31;
  const int wave = tid >> 5;
  const int wm   = wave >> 2;               // 0..1 -> 64 rows each
  const int wn   = wave & 3;                // 0..3 -> 32 cols each
  const int krow = lane & 15;               // row/col within 16-tile
  const int koff = (lane & 16) ? 8 : 0;     // ISA 16-bit fragment K base

  v8f acc[4][2] = {};

  #pragma unroll
  for (int kc = 0; kc < D_DIM; kc += 32) {
    v16bf ahi[4], alo[4], bhi[2], blo[2];
    #pragma unroll
    for (int mt = 0; mt < 4; ++mt) {
      const int row = (wm * 64 + mt * 16 + krow) * KP + kc + koff;
      ahi[mt] = load_frag(&sm.xhi[row]);
      alo[mt] = load_frag(&sm.xlo[row]);
    }
    #pragma unroll
    for (int nt = 0; nt < 2; ++nt) {
      const int col = (wn * 32 + nt * 16 + krow) * KP + kc + koff;
      bhi[nt] = load_frag(&sm.yhi[col]);
      blo[nt] = load_frag(&sm.ylo[col]);
    }
    #pragma unroll
    for (int mt = 0; mt < 4; ++mt) {
      #pragma unroll
      for (int nt = 0; nt < 2; ++nt) {
        acc[mt][nt] = __builtin_amdgcn_wmma_f32_16x16x32_bf16(
            false, ahi[mt], false, bhi[nt], (short)0, acc[mt][nt], false, false);
        acc[mt][nt] = __builtin_amdgcn_wmma_f32_16x16x32_bf16(
            false, ahi[mt], false, blo[nt], (short)0, acc[mt][nt], false, false);
        acc[mt][nt] = __builtin_amdgcn_wmma_f32_16x16x32_bf16(
            false, alo[mt], false, bhi[nt], (short)0, acc[mt][nt], false, false);
      }
    }
  }

  // ---- Phase 3: fused epilogue + store ----
  // C/D layout: VGPR v: lanes 0-15 -> M=v, lanes 16-31 -> M=v+8; N = lane&15.
  float* outb = out + (size_t)b * (size_t)N * (size_t)M;
  const int lrow_off = (lane & 16) >> 1;    // 0 or 8
  const int ncol     = lane & 15;
  #pragma unroll
  for (int mt = 0; mt < 4; ++mt) {
    #pragma unroll
    for (int nt = 0; nt < 2; ++nt) {
      #pragma unroll
      for (int v = 0; v < 8; ++v) {
        const int lr = wm * 64 + mt * 16 + v + lrow_off;  // local n
        const int lc = wn * 32 + nt * 16 + ncol;          // local m
        const float dot = acc[mt][nt][v];
        const float sqd = fmaxf(sm.x2[lr] + sm.y2[lc] - 2.0f * dot, 0.0f);
        const float md  = sm.mx[lr] - sm.my[lc];
        const float val = __cosf(md) * __expf(-0.5f * sqd);
        outb[((size_t)ntile * TILE + lr) * (size_t)M +
             ((size_t)mtile * TILE + lc)] = val;
      }
    }
  }
}

extern "C" void kernel_launch(void* const* d_in, const int* in_sizes, int n_in,
                              void* d_out, int out_size, void* d_ws, size_t ws_size,
                              hipStream_t stream) {
  const float* x     = (const float*)d_in[0];   // [B,N,D]
  const float* y     = (const float*)d_in[1];   // [B,M,D]
  const float* mu    = (const float*)d_in[2];   // [D,1]
  const float* logs2 = (const float*)d_in[3];   // [D]
  float* out = (float*)d_out;                   // [B,N,M] f32

  const int D = in_sizes[3];                    // 64
  const int B = 4;
  const int N = in_sizes[0] / (B * D);          // 4096
  const int M = in_sizes[1] / (B * D);          // 4096

  dim3 grid((unsigned)(M / TILE), (unsigned)(N / TILE), (unsigned)B);
  cosgauss_wmma_kernel<<<grid, 256, 0, stream>>>(x, y, mu, logs2, out, N, M);
}
